// FeatExtract_45449343926482
// MI455X (gfx1250) — compile-verified, matrix operand order checked by reading, not run on hardware
//
#include <hip/hip_runtime.h>

typedef __bf16 bf16_t;
typedef __attribute__((ext_vector_type(16))) __bf16 v16bf;
typedef __attribute__((ext_vector_type(8)))  float  v8f;
typedef __attribute__((ext_vector_type(4)))  float  v4f;

#define NN 50000
#define NE 800000
#define D 128
#define EPS 1e-5f
#define SROW 132   // padded LDS row stride (floats) to avoid 64-bank conflicts

__device__ __forceinline__ float gelu_exact(float x) {
    return 0.5f * x * (1.0f + erff(x * 0.70710678118654752f));
}

// ---------------------------------------------------------------------------
// Pre-pack 9 fp32 128x128 weight matrices (3 blocks x {Wn, We, Wo}) into bf16
// WMMA B-fragments: frag(kt 0..3, nt 0..7) -> 32 lanes x 16 contiguous bf16.
// Element (lane, j) holds W[kt*32 + kmap(lane,j)][nt*16 + (lane&15)], where
// kmap matches the ISA 16-bit operand layout (lanes 0-15: K 0-7,16-23;
// lanes 16-31: K 8-15,24-31).
// ---------------------------------------------------------------------------
__global__ void prepack_kernel(const float* __restrict__ Wn,
                               const float* __restrict__ We,
                               const float* __restrict__ Wo,
                               bf16_t* __restrict__ out) {
    int t = blockIdx.x * blockDim.x + threadIdx.x;
    if (t >= 9 * 16384) return;
    int mat  = t / 16384;          // blk*3 + which
    int f    = t & 16383;
    int j    = f & 15;
    int lane = (f >> 4) & 31;
    int nt   = (f >> 9) & 7;
    int kt   = (f >> 12) & 3;
    int blk = mat / 3, which = mat % 3;
    const float* W = (which == 0) ? Wn : (which == 1) ? We : Wo;
    W += (size_t)blk * D * D;
    int k = kt * 32 + ((lane >> 4) ? 8 : 0) + ((j >> 3) ? 16 : 0) + (j & 7);
    int n = nt * 16 + (lane & 15);
    out[t] = (bf16_t)W[k * D + n];
}

__global__ void zero_kernel(float* __restrict__ p, size_t n) {
    size_t i = (size_t)blockIdx.x * blockDim.x + threadIdx.x;
    if (i < n) p[i] = 0.0f;
}

__device__ __forceinline__ v16bf pack_a_frag(v4f a, v4f b, v4f c, v4f d) {
    v16bf r;
#pragma unroll
    for (int j = 0; j < 4; ++j) {
        r[j]      = (bf16_t)a[j];
        r[4 + j]  = (bf16_t)b[j];
        r[8 + j]  = (bf16_t)c[j];
        r[12 + j] = (bf16_t)d[j];
    }
    return r;
}

// ---------------------------------------------------------------------------
// Fused  dst = prev + alpha * LN(GELU((X0 + beta*X1) @ W)) * gamma + bias
// 256 threads (8 waves), 128 rows per workgroup, WMMA bf16 K=32 core.
// FUSE_IN / FUSE_OUT are compile-time so every load path stays b128-vector.
// ---------------------------------------------------------------------------
template <bool FUSE_IN, bool FUSE_OUT>
__global__ void __launch_bounds__(256)
gemm_gelu_ln_kernel(const float* __restrict__ X0,
                    const float* __restrict__ X1,
                    const float* __restrict__ beta_p,
                    const bf16_t* __restrict__ Wp,
                    const float* __restrict__ gamma,
                    const float* __restrict__ bias,
                    const float* __restrict__ prev,
                    const float* __restrict__ alpha_p,
                    float* __restrict__ dst,
                    int nrows) {
    __shared__ float sC[128 * SROW];
    const int tid  = threadIdx.x;
    const int wave = tid >> 5, lane = tid & 31;
    const int m = lane & 15, hi = lane >> 4;
    const int row0 = blockIdx.x * 128;

    int rowA = row0 + wave * 16 + m;
    if (rowA >= nrows) rowA = nrows - 1;     // clamp; result discarded

    const float beta = FUSE_IN ? *beta_p : 0.0f;
    const v4f bv = {beta, beta, beta, beta};
    const float* xr  = X0 + (size_t)rowA * D;
    const float* xr1 = FUSE_IN ? X1 + (size_t)rowA * D : nullptr;

    v8f acc[8] = {};
    for (int kt = 0; kt < 4; ++kt) {
        const int kb = kt * 32 + hi * 8;
        v4f xa = *(const v4f*)(xr + kb);
        v4f xb = *(const v4f*)(xr + kb + 4);
        v4f xc = *(const v4f*)(xr + kb + 16);
        v4f xd = *(const v4f*)(xr + kb + 20);
        if (FUSE_IN) {
            xa += bv * *(const v4f*)(xr1 + kb);
            xb += bv * *(const v4f*)(xr1 + kb + 4);
            xc += bv * *(const v4f*)(xr1 + kb + 16);
            xd += bv * *(const v4f*)(xr1 + kb + 20);
        }
        v16bf a = pack_a_frag(xa, xb, xc, xd);
#pragma unroll
        for (int nt = 0; nt < 8; ++nt) {
            v16bf b = *(const v16bf*)(Wp + (((size_t)(kt * 8 + nt) * 32 + lane) << 4));
            acc[nt] = __builtin_amdgcn_wmma_f32_16x16x32_bf16(
                false, a, false, b, (short)0, acc[nt], false, false);
        }
    }
    // C layout: vgpr r, lanes 0-15 -> M=r, lanes 16-31 -> M=r+8, N=lane&15
#pragma unroll
    for (int nt = 0; nt < 8; ++nt)
#pragma unroll
        for (int r = 0; r < 8; ++r)
            sC[(wave * 16 + r + hi * 8) * SROW + nt * 16 + m] = acc[nt][r];
    __syncthreads();

    // Epilogue: 2 threads per row (64 cols each), GELU -> LayerNorm -> residual
    const int row = tid >> 1, half = tid & 1;
    const int rg  = row0 + row;
    float* rp = &sC[row * SROW + half * 64];
    float s = 0.0f, s2 = 0.0f;
#pragma unroll 8
    for (int c = 0; c < 64; ++c) {
        float g = gelu_exact(rp[c]);
        rp[c] = g;
        s += g; s2 += g * g;
    }
    s  += __shfl_xor(s, 1, 32);
    s2 += __shfl_xor(s2, 1, 32);
    const float mu  = s * (1.0f / 128.0f);
    const float var = s2 * (1.0f / 128.0f) - mu * mu;
    const float rs  = rsqrtf(var + EPS);

    if (rg < nrows) {
        const float al = FUSE_OUT ? *alpha_p : 1.0f;
        const v4f av = {al, al, al, al};
        const size_t base = (size_t)rg * D + half * 64;
        const float* gp = gamma + half * 64;
        const float* bp = bias + half * 64;
#pragma unroll
        for (int c4 = 0; c4 < 16; ++c4) {
            v4f v;
#pragma unroll
            for (int j = 0; j < 4; ++j)
                v[j] = (rp[c4 * 4 + j] - mu) * rs * gp[c4 * 4 + j] + bp[c4 * 4 + j];
            if (FUSE_OUT) {
                v4f pv = *(const v4f*)(prev + base + c4 * 4);
                v = pv + av * v;
            }
            *(v4f*)(dst + base + c4 * 4) = v;
        }
    }
}

// ---------------------------------------------------------------------------
// Fused per-edge:  he = exp(LN(E @ We));  atomicAdd(h[dst], hv[src] * he)
// he never hits global memory; edge_feats streamed with NT hint (bigger than
// L2, read once per block) so it doesn't evict the L2-resident hv/h tensors.
// ---------------------------------------------------------------------------
__global__ void __launch_bounds__(256)
edge_gemm_scatter_kernel(const float* __restrict__ EF,
                         const bf16_t* __restrict__ Wp,
                         const float* __restrict__ gamma,
                         const float* __restrict__ bias,
                         const int* __restrict__ src,
                         const int* __restrict__ dsti,
                         const float* __restrict__ hv,
                         float* __restrict__ h) {
    __shared__ float sC[128 * SROW];
    const int tid  = threadIdx.x;
    const int wave = tid >> 5, lane = tid & 31;
    const int m = lane & 15, hi = lane >> 4;
    const int row0 = blockIdx.x * 128;        // NE % 128 == 0, no guard needed

    const float* xr = EF + (size_t)(row0 + wave * 16 + m) * D;

    v8f acc[8] = {};
    for (int kt = 0; kt < 4; ++kt) {
        const int kb = kt * 32 + hi * 8;
        v4f xa = __builtin_nontemporal_load((const v4f*)(xr + kb));
        v4f xb = __builtin_nontemporal_load((const v4f*)(xr + kb + 4));
        v4f xc = __builtin_nontemporal_load((const v4f*)(xr + kb + 16));
        v4f xd = __builtin_nontemporal_load((const v4f*)(xr + kb + 20));
        v16bf a = pack_a_frag(xa, xb, xc, xd);
#pragma unroll
        for (int nt = 0; nt < 8; ++nt) {
            v16bf b = *(const v16bf*)(Wp + (((size_t)(kt * 8 + nt) * 32 + lane) << 4));
            acc[nt] = __builtin_amdgcn_wmma_f32_16x16x32_bf16(
                false, a, false, b, (short)0, acc[nt], false, false);
        }
    }
#pragma unroll
    for (int nt = 0; nt < 8; ++nt)
#pragma unroll
        for (int r = 0; r < 8; ++r)
            sC[(wave * 16 + r + hi * 8) * SROW + nt * 16 + m] = acc[nt][r];
    __syncthreads();

    const int row = tid >> 1, half = tid & 1;
    const int e   = row0 + row;
    float* rp = &sC[row * SROW + half * 64];
    float s = 0.0f, s2 = 0.0f;
#pragma unroll 8
    for (int c = 0; c < 64; ++c) { float v = rp[c]; s += v; s2 += v * v; }
    s  += __shfl_xor(s, 1, 32);
    s2 += __shfl_xor(s2, 1, 32);
    const float mu  = s * (1.0f / 128.0f);
    const float var = s2 * (1.0f / 128.0f) - mu * mu;
    const float rs  = rsqrtf(var + EPS);

    const int sN = src[e], dN = dsti[e];
    const float* hvp = hv + (size_t)sN * D + half * 64;
    float* hp = h + (size_t)dN * D + half * 64;
    const float* gp = gamma + half * 64;
    const float* bp = bias + half * 64;
#pragma unroll 8
    for (int c = 0; c < 64; ++c) {
        float he = expf((rp[c] - mu) * rs * gp[c] + bp[c]);
        atomicAdd(&hp[c], hvp[c] * he);
    }
}

// ---------------------------------------------------------------------------
extern "C" void kernel_launch(void* const* d_in, const int* in_sizes, int n_in,
                              void* d_out, int out_size, void* d_ws, size_t ws_size,
                              hipStream_t stream) {
    const float* node_feats = (const float*)d_in[0];
    const float* edge_feats = (const float*)d_in[1];
    const int*   src        = (const int*)d_in[2];
    const int*   dst        = (const int*)d_in[3];
    const float* We         = (const float*)d_in[4];
    const float* ln_e_g     = (const float*)d_in[5];
    const float* ln_e_b     = (const float*)d_in[6];
    const float* Wn         = (const float*)d_in[7];
    const float* ln_n_g     = (const float*)d_in[8];
    const float* ln_n_b     = (const float*)d_in[9];
    const float* Wo         = (const float*)d_in[10];
    const float* ln_o_g     = (const float*)d_in[11];
    const float* ln_o_b     = (const float*)d_in[12];
    const float* alphas     = (const float*)d_in[13];
    const float* beltas     = (const float*)d_in[14];
    float* out = (float*)d_out;

    char* ws = (char*)d_ws;
    bf16_t* Wpack = (bf16_t*)ws;                        // 9*16384*2 = 294912 B
    float* hv   = (float*)(ws + 294912);                // 6.4M floats
    float* hbuf = hv + (size_t)NN * D;                  // 6.4M floats
    float* bufA = hbuf + (size_t)NN * D;                // 6.4M floats

    auto Wn_p = [&](int i) { return Wpack + (size_t)(i * 3 + 0) * 16384; };
    auto We_p = [&](int i) { return Wpack + (size_t)(i * 3 + 1) * 16384; };
    auto Wo_p = [&](int i) { return Wpack + (size_t)(i * 3 + 2) * 16384; };

    const int NODE_BLK = (NN + 127) / 128;  // 391
    const int EDGE_BLK = NE / 128;          // 6250
    const size_t HN = (size_t)NN * D;
    const int ZB = (int)((HN + 255) / 256);

    prepack_kernel<<<(9 * 16384 + 255) / 256, 256, 0, stream>>>(Wn, We, Wo, Wpack);

    // ---- block 0: b0 = blk(0, node_feats) -> bufA ----
    gemm_gelu_ln_kernel<false, false><<<NODE_BLK, 256, 0, stream>>>(
        node_feats, nullptr, nullptr, Wn_p(0), ln_n_g, ln_n_b,
        nullptr, nullptr, hv, NN);
    zero_kernel<<<ZB, 256, 0, stream>>>(hbuf, HN);
    edge_gemm_scatter_kernel<<<EDGE_BLK, 256, 0, stream>>>(
        edge_feats, We_p(0), ln_e_g, ln_e_b, src, dst, hv, hbuf);
    gemm_gelu_ln_kernel<false, false><<<NODE_BLK, 256, 0, stream>>>(
        hbuf, nullptr, nullptr, Wo_p(0), ln_o_g, ln_o_b,
        nullptr, nullptr, bufA, NN);

    // ---- block 1: r1 = bufA + a1*blk(1, bufA + b1*node_feats) -> out ----
    gemm_gelu_ln_kernel<true, false><<<NODE_BLK, 256, 0, stream>>>(
        bufA, node_feats, beltas + 1, Wn_p(1), ln_n_g + D, ln_n_b + D,
        nullptr, nullptr, hv, NN);
    zero_kernel<<<ZB, 256, 0, stream>>>(hbuf, HN);
    edge_gemm_scatter_kernel<<<EDGE_BLK, 256, 0, stream>>>(
        edge_feats, We_p(1), ln_e_g + D, ln_e_b + D, src, dst, hv, hbuf);
    gemm_gelu_ln_kernel<false, true><<<NODE_BLK, 256, 0, stream>>>(
        hbuf, nullptr, nullptr, Wo_p(1), ln_o_g + D, ln_o_b + D,
        bufA, alphas + 1, out, NN);

    // ---- block 2: r2 = out + a2*blk(2, out + b2*node_feats) -> out ----
    gemm_gelu_ln_kernel<true, false><<<NODE_BLK, 256, 0, stream>>>(
        out, node_feats, beltas + 2, Wn_p(2), ln_n_g + 2 * D, ln_n_b + 2 * D,
        nullptr, nullptr, hv, NN);
    zero_kernel<<<ZB, 256, 0, stream>>>(hbuf, HN);
    edge_gemm_scatter_kernel<<<EDGE_BLK, 256, 0, stream>>>(
        edge_feats, We_p(2), ln_e_g + 2 * D, ln_e_b + 2 * D, src, dst, hv, hbuf);
    gemm_gelu_ln_kernel<false, true><<<NODE_BLK, 256, 0, stream>>>(
        hbuf, nullptr, nullptr, Wo_p(2), ln_o_g + 2 * D, ln_o_b + 2 * D,
        out, alphas + 2, out, NN);
}
